// CircleLoss_63161789054983
// MI455X (gfx1250) — matrix operand order repaired and checked by reading, not compiled, not run on hardware
//
#include <hip/hip_runtime.h>
#include <math.h>

#define NROWS 8192
#define KDIM  128

typedef __attribute__((ext_vector_type(16))) _Float16 v16h;
typedef __attribute__((ext_vector_type(8)))  _Float16 v8h;
typedef __attribute__((ext_vector_type(8)))  float    v8f;

// ---------------------------------------------------------------------------
// Kernel 1: per-row norms, f16 hi/lo split, and diagonal (positive-pair)
// exp-sum accumulated in double (logit_p can reach 252 -> needs double range).
// One 128-thread block per row; fully coalesced.
// ---------------------------------------------------------------------------
__global__ __launch_bounds__(128) void prep_kernel(
    const float* __restrict__ enc, const float* __restrict__ dec,
    _Float16* __restrict__ a_hi, _Float16* __restrict__ a_lo,
    _Float16* __restrict__ b_hi, _Float16* __restrict__ b_lo,
    float* __restrict__ en, float* __restrict__ dn,
    double* __restrict__ diag_acc)
{
    const int row = blockIdx.x;
    const int k   = threadIdx.x;
    const size_t idx = (size_t)row * KDIM + k;

    const float a = enc[idx];
    const float b = dec[idx];

    const _Float16 ah = (_Float16)a;
    const _Float16 al = (_Float16)(a - (float)ah);
    const _Float16 bh = (_Float16)b;
    const _Float16 bl = (_Float16)(b - (float)bh);
    a_hi[idx] = ah; a_lo[idx] = al;
    b_hi[idx] = bh; b_lo[idx] = bl;

    __shared__ float r_aa[128], r_bb[128], r_ab[128];
    r_aa[k] = a * a; r_bb[k] = b * b; r_ab[k] = a * b;
    __syncthreads();
    #pragma unroll
    for (int off = 64; off > 0; off >>= 1) {
        if (k < off) {
            r_aa[k] += r_aa[k + off];
            r_bb[k] += r_bb[k + off];
            r_ab[k] += r_ab[k + off];
        }
        __syncthreads();
    }
    if (k == 0) {
        const float ea = sqrtf(r_aa[0]);
        const float eb = sqrtf(r_bb[0]);
        en[row] = ea; dn[row] = eb;
        const float sp = r_ab[0] / (ea * eb + 1e-5f);
        const float ap = fmaxf(1.25f - sp, 0.0f);
        const float lp = -ap * (sp - 0.75f) * 64.0f;   // logit_p in [-4, 252]
        atomicAdd(diag_acc, exp((double)lp));
    }
}

// ---------------------------------------------------------------------------
// WMMA fragment loaders (layouts per CDNA5 ISA 7.12.2):
// A 16x32 f16: lane L -> row M=L%16; halves 0-7 = K[base..base+7],
//              halves 8-15 = K[base+16..base+23], base = ks + (L>=16 ? 8 : 0)
// B 32x16 f16: lane L -> col N=L%16; halves 0-7 = K[kb..kb+7],
//              halves 8-15 = K[kb+16..kb+23], kb = ks + (L>=16 ? 8 : 0)
// ---------------------------------------------------------------------------
__device__ __forceinline__ v16h load_fragA(const _Float16* __restrict__ p) {
    union { v16h v; v8h h[2]; } u;
    u.h[0] = *(const v8h*)(p);
    u.h[1] = *(const v8h*)(p + 16);
    return u.v;
}
__device__ __forceinline__ v16h load_fragB(const _Float16* __restrict__ p) {
    union { v16h v; v8h h[2]; } u;
    u.h[0] = *(const v8h*)(p);
    u.h[1] = *(const v8h*)(p + 16);
    return u.v;
}
__device__ __forceinline__ v8f wmma_f16(v16h a, v16h b, v8f c) {
    return __builtin_amdgcn_wmma_f32_16x16x32_f16(
        false, a, false, b, (short)0, c, false, false);
}

// ---------------------------------------------------------------------------
// Kernel 2: fused cosine-sim GEMM + circle-loss epilogue + per-diagonal
// exp-sum binning. 128x128 output tile per block (8 waves, 32x64 per wave).
// dot = Ah*Bh + Ah*Bl + Al*Bh  (split-f16 ~ fp32 accuracy), fp32 accumulate.
// ---------------------------------------------------------------------------
__global__ __launch_bounds__(256) void gemm_loss_kernel(
    const _Float16* __restrict__ a_hi, const _Float16* __restrict__ a_lo,
    const _Float16* __restrict__ b_hi, const _Float16* __restrict__ b_lo,
    const float* __restrict__ en, const float* __restrict__ dn,
    float* __restrict__ diag_sum)
{
    const int tile_m = blockIdx.y * 128;
    const int tile_n = blockIdx.x * 128;
    const int wave = threadIdx.x >> 5;
    const int lane = threadIdx.x & 31;
    const int m0 = tile_m + 32 * (wave & 3);   // wave's 32-row patch
    const int n0 = tile_n + 64 * (wave >> 2);  // wave's 64-col patch

    __shared__ float lds_bucket[256];          // 255 diagonal bins used
    for (int i = threadIdx.x; i < 256; i += 256) lds_bucket[i] = 0.0f;
    __syncthreads();

    const int lr    = lane & 15;
    const int akoff = (lane >> 4) * 8;
    const int bkoff = (lane >> 4) * 8;

    v8f acc[2][4];
    const v8f vzero = {0.f, 0.f, 0.f, 0.f, 0.f, 0.f, 0.f, 0.f};
    #pragma unroll
    for (int i = 0; i < 2; ++i)
        #pragma unroll
        for (int j = 0; j < 4; ++j) acc[i][j] = vzero;

    #pragma unroll
    for (int ks = 0; ks < KDIM; ks += 32) {
        v16h Ah[2], Al[2], Bh[4], Bl[4];
        #pragma unroll
        for (int i = 0; i < 2; ++i) {
            const size_t off = (size_t)(m0 + 16 * i + lr) * KDIM + ks + akoff;
            Ah[i] = load_fragA(a_hi + off);
            Al[i] = load_fragA(a_lo + off);
        }
        #pragma unroll
        for (int j = 0; j < 4; ++j) {
            const size_t off = (size_t)(n0 + 16 * j + lr) * KDIM + ks + bkoff;
            Bh[j] = load_fragB(b_hi + off);
            Bl[j] = load_fragB(b_lo + off);
        }
        #pragma unroll
        for (int i = 0; i < 2; ++i)
            #pragma unroll
            for (int j = 0; j < 4; ++j) {
                acc[i][j] = wmma_f16(Ah[i], Bh[j], acc[i][j]);
                acc[i][j] = wmma_f16(Ah[i], Bl[j], acc[i][j]);
                acc[i][j] = wmma_f16(Al[i], Bh[j], acc[i][j]);
            }
    }

    // Epilogue. C/D layout: VGPR r -> M = r + (lane>=16 ? 8 : 0); N = lane%16.
    const int rsel = (lane >> 4) * 8;
    #pragma unroll
    for (int i = 0; i < 2; ++i) {
        const int mb = m0 + 16 * i + rsel;
        float env[8];
        #pragma unroll
        for (int r = 0; r < 8; ++r) env[r] = en[mb + r];
        #pragma unroll
        for (int j = 0; j < 4; ++j) {
            const int n = n0 + 16 * j + lr;
            const float dnj = dn[n];
            #pragma unroll
            for (int r = 0; r < 8; ++r) {
                const int m = mb + r;
                const float s   = acc[i][j][r] / (env[r] * dnj + 1e-5f);
                const float anv = fmaxf(s + 0.25f, 0.0f);
                // logit_n = gamma*an*(s-0.25) in [-4,60]; exp fits fp32.
                // an==0 still contributes exp(0)=1. Main diagonal excluded.
                const float w = (m == n) ? 0.0f
                                         : __expf(64.0f * anv * (s - 0.25f));
                const int od = (n - m) - (tile_n - tile_m) + 127;  // 0..254
                atomicAdd(&lds_bucket[od], w);
            }
        }
    }
    __syncthreads();

    if (threadIdx.x < 255) {
        const float v = lds_bucket[threadIdx.x];
        const int d = (tile_n - tile_m) + ((int)threadIdx.x - 127);
        const int o = d & (NROWS - 1);             // wrap mod 8192
        if (o != 0) atomicAdd(&diag_sum[o], v);
    }
}

// ---------------------------------------------------------------------------
// Kernel 3: result = mean_{o=1..N-1} softplus(log(diag_sum[o]) + log(diag_acc))
// ---------------------------------------------------------------------------
__global__ __launch_bounds__(256) void finalize_kernel(
    const float* __restrict__ diag_sum, const double* __restrict__ diag_acc,
    float* __restrict__ out)
{
    __shared__ double red[256];
    const double lse_p = log(diag_acc[0]);
    double local = 0.0;
    for (int o = 1 + (int)threadIdx.x; o < NROWS; o += 256) {
        const double x = log((double)diag_sum[o]) + lse_p;
        local += (x > 30.0) ? x : log1p(exp(x));
    }
    red[threadIdx.x] = local;
    __syncthreads();
    #pragma unroll
    for (int off = 128; off > 0; off >>= 1) {
        if ((int)threadIdx.x < off) red[threadIdx.x] += red[threadIdx.x + off];
        __syncthreads();
    }
    if (threadIdx.x == 0) out[0] = (float)(red[0] / (double)(NROWS - 1));
}

// ---------------------------------------------------------------------------
extern "C" void kernel_launch(void* const* d_in, const int* in_sizes, int n_in,
                              void* d_out, int out_size, void* d_ws, size_t ws_size,
                              hipStream_t stream)
{
    const float* enc = (const float*)d_in[0];
    const float* dec = (const float*)d_in[1];

    char* p = (char*)d_ws;
    const size_t half_mat = (size_t)NROWS * KDIM * sizeof(_Float16); // 2 MB
    _Float16* a_hi = (_Float16*)p; p += half_mat;
    _Float16* a_lo = (_Float16*)p; p += half_mat;
    _Float16* b_hi = (_Float16*)p; p += half_mat;
    _Float16* b_lo = (_Float16*)p; p += half_mat;
    float* en       = (float*)p;   p += NROWS * sizeof(float);
    float* dn       = (float*)p;   p += NROWS * sizeof(float);
    float* diag_sum = (float*)p;   p += NROWS * sizeof(float);
    double* diag_acc = (double*)p;

    // Zero the accumulators (diag_sum + diag_acc are contiguous).
    hipMemsetAsync(diag_sum, 0, NROWS * sizeof(float) + sizeof(double), stream);

    prep_kernel<<<NROWS, 128, 0, stream>>>(enc, dec, a_hi, a_lo, b_hi, b_lo,
                                           en, dn, diag_acc);

    dim3 grid(NROWS / 128, NROWS / 128);   // 64 x 64 tiles
    gemm_loss_kernel<<<grid, 256, 0, stream>>>(a_hi, a_lo, b_hi, b_lo,
                                               en, dn, diag_sum);

    finalize_kernel<<<1, 256, 0, stream>>>(diag_sum, diag_acc, (float*)d_out);
}
